// GGN_IO_B_35983236006288
// MI455X (gfx1250) — matrix operand order repaired and checked by reading, not compiled, not run on hardware
//
#include <hip/hip_runtime.h>

typedef __attribute__((ext_vector_type(16))) _Float16 v16h;
typedef __attribute__((ext_vector_type(8)))  float    v8f;
typedef int v4i_vs __attribute__((vector_size(16)));   // matches builtin param type
typedef __attribute__((address_space(1))) v4i_vs* g128p;
typedef __attribute__((address_space(3))) v4i_vs* l128p;

#define NODES 1000
#define DIM   128
#define HID   256
#define MTILE 128
#define NTILES 8          // ceil(1000/128)

// padded LDS strides (in halfs) to avoid 256B/512B bank-stride pathologies
#define WA_STRIDE 136     // 128 + 8
#define W2_STRIDE 264     // 256 + 8
#define H1_STRIDE 264

#define SWA_OFF   0                         // 256*136*2 = 69632
#define SW2_OFF   69632                     // 256*264*2 = 135168
#define SH1_OFF   204800                    // 8*16*264*2 = 67584
#define SACC_OFF  272384                    // 256*4 = 1024
#define SEB_OFF   273408                    // 256*4 = 1024  (ebias row)
#define SB2_OFF   274432                    // 256*4 = 1024  (b_e2e row)
#define SMEM_BYTES 275456                   // <= 320KB WGP LDS

#if __has_builtin(__builtin_amdgcn_global_load_async_to_lds_b128)
#define HAVE_ASYNC_LDS 1
#else
#define HAVE_ASYNC_LDS 0
#endif

// ---- fragment loaders per CDNA5 ISA 7.12.2 layouts (wave32) ----

// A 16x32 f16: lanes 0-15 hold M=lane, K={k0..k0+7, k0+16..k0+23};
//              lanes 16-31 hold M=lane-16, K={k0+8.., k0+24..}
__device__ inline v16h afrag_lds(const _Float16* base, int stride, int row, int k0, int lane) {
  const int kb = (lane & 16) ? 8 : 0;
  const _Float16* p = base + row * stride + k0 + kb;
  union { uint4 q[2]; v16h v; } u;
  u.q[0] = *(const uint4*)(p);
  u.q[1] = *(const uint4*)(p + 16);
  return u.v;
}

// B 32x16 f16: lane holds col = lane&15; lanes 0-15: K={k0..k0+7, k0+16..k0+23}, lanes 16-31: K={k0+8.., k0+24..}
__device__ inline v16h bfrag_lds(const _Float16* base, int stride, int col, int k0, int lane) {
  const int kb = (lane & 16) ? 8 : 0;
  const _Float16* p = base + col * stride + k0 + kb;
  union { uint4 q[2]; v16h v; } u;
  u.q[0] = *(const uint4*)(p);
  u.q[1] = *(const uint4*)(p + 16);
  return u.v;
}

// A fragment straight from a global f32 row, converting to f16
__device__ inline v16h afrag_gf32(const float* rowptr, int k0, int lane) {
  const int kb = (lane & 16) ? 8 : 0;
  const float4* p = (const float4*)(rowptr + k0 + kb);
  float4 f0 = p[0], f1 = p[1];       // K = k0+kb .. +7
  const float4* q = (const float4*)(rowptr + k0 + kb + 16);
  float4 f2 = q[0], f3 = q[1];       // K = k0+kb+16 .. +23
  v16h a;
  a[0]=(_Float16)f0.x; a[1]=(_Float16)f0.y; a[2]=(_Float16)f0.z; a[3]=(_Float16)f0.w;
  a[4]=(_Float16)f1.x; a[5]=(_Float16)f1.y; a[6]=(_Float16)f1.z; a[7]=(_Float16)f1.w;
  a[8]=(_Float16)f2.x; a[9]=(_Float16)f2.y; a[10]=(_Float16)f2.z; a[11]=(_Float16)f2.w;
  a[12]=(_Float16)f3.x; a[13]=(_Float16)f3.y; a[14]=(_Float16)f3.z; a[15]=(_Float16)f3.w;
  return a;
}

// copy one 16B chunk global->LDS (async path writes LDS directly, no VGPR bounce)
__device__ inline void stage16(const _Float16* gsrc, _Float16* ldst) {
#if HAVE_ASYNC_LDS
  __builtin_amdgcn_global_load_async_to_lds_b128(
      (g128p)gsrc, (l128p)ldst, 0, 0);
#else
  *(uint4*)ldst = *(const uint4*)gsrc;
#endif
}

__device__ inline void stage_wait() {
#if HAVE_ASYNC_LDS
#if __has_builtin(__builtin_amdgcn_s_wait_asynccnt)
  __builtin_amdgcn_s_wait_asynccnt(0);
#else
  asm volatile("s_wait_asynccnt 0x0" ::: "memory");
#endif
#endif
}

// ---- prep: f32 -> f16 weight conversion ----
__global__ void __launch_bounds__(256) k_convert(const float* __restrict__ Wn2e,
                                                 const float* __restrict__ We2e,
                                                 _Float16* __restrict__ wah,
                                                 _Float16* __restrict__ w2h) {
  int idx = blockIdx.x * 256 + threadIdx.x;
  if (idx < HID * DIM) {               // wah[o][k] = Wn2e[o][k], k<128 ("starter" half)
    int o = idx >> 7, k = idx & 127;
    wah[idx] = (_Float16)Wn2e[o * (2 * DIM) + k];
  }
  if (idx < HID * HID) {
    w2h[idx] = (_Float16)We2e[idx];
  }
}

// ---- prep: per-batch ender bias  e[b] = b_n2e + Wn2e[:,128:256] @ x[b,i]; zero accum ----
__global__ void __launch_bounds__(256) k_ebias(const float* __restrict__ x,
                                               const float* __restrict__ Wn2e,
                                               const float* __restrict__ bn2e,
                                               const int* __restrict__ ip,
                                               float* __restrict__ ebias,
                                               float* __restrict__ accum) {
  __shared__ float xi[DIM];
  const int b = blockIdx.x, tid = threadIdx.x;
  const int i = *ip;
  if (tid < DIM) xi[tid] = x[((size_t)b * NODES + i) * DIM + tid];
  __syncthreads();
  float s = bn2e[tid];
  const float* w = Wn2e + tid * (2 * DIM) + DIM;
  for (int k = 0; k < DIM; ++k) s += w[k] * xi[k];
  ebias[b * HID + tid] = s;
  accum[b * HID + tid] = 0.0f;
}

// ---- main WMMA kernel: L1 + L2 + adj-weighted node reduction ----
__global__ void __launch_bounds__(256) k_main(const float* __restrict__ x,
                                              const float* __restrict__ adj,
                                              const _Float16* __restrict__ wah,
                                              const _Float16* __restrict__ w2h,
                                              const float* __restrict__ ebias,
                                              const float* __restrict__ be2e,
                                              float* __restrict__ accum) {
  extern __shared__ char smem[];
  _Float16* sWa  = (_Float16*)(smem + SWA_OFF);   // [256][WA_STRIDE]
  _Float16* sW2  = (_Float16*)(smem + SW2_OFF);   // [256][W2_STRIDE]
  _Float16* sH1  = (_Float16*)(smem + SH1_OFF);   // [8 waves][16][H1_STRIDE]
  float*    sAcc = (float*)(smem + SACC_OFF);     // [256]
  float*    sEb  = (float*)(smem + SEB_OFF);      // [256]
  float*    sB2  = (float*)(smem + SB2_OFF);      // [256]

  const int b     = blockIdx.y;
  const int ntile = blockIdx.x;
  const int tid   = threadIdx.x;
  const int wave  = tid >> 5;
  const int lane  = tid & 31;
  const int lc    = lane & 15;                    // col / row-in-wave selector

  // stage weights into LDS (padded strides; async global->LDS when available)
  for (int idx = tid; idx < (HID * DIM) / 8; idx += 256) {
    int o = idx >> 4, kv = idx & 15;
    stage16(wah + idx * 8, sWa + o * WA_STRIDE + kv * 8);
  }
  for (int idx = tid; idx < (HID * HID) / 8; idx += 256) {
    int o = idx >> 5, kv = idx & 31;
    stage16(w2h + idx * 8, sW2 + o * W2_STRIDE + kv * 8);
  }
  sAcc[tid] = 0.0f;
  sEb[tid]  = ebias[b * HID + tid];
  sB2[tid]  = be2e[tid];
  stage_wait();
  __syncthreads();

  // ---- layer 1: h1 = relu(Wa @ x_n + e[b]) for this wave's 16 node rows ----
  v16h ax[4];
  {
    int n  = ntile * MTILE + wave * 16 + lc;
    int nc = n < NODES ? n : NODES - 1;           // clamped rows get adj-weight 0 below
    const float* rowptr = x + ((size_t)b * NODES + nc) * DIM;
#pragma unroll
    for (int kk = 0; kk < 4; ++kk) ax[kk] = afrag_gf32(rowptr, kk * 32, lane);
  }
  _Float16* myH1 = sH1 + wave * 16 * H1_STRIDE;

#pragma unroll 2
  for (int t = 0; t < 16; ++t) {
    const int col = t * 16 + lc;
    const float e = sEb[col];
    v8f acc;
#pragma unroll
    for (int r = 0; r < 8; ++r) acc[r] = e;
#pragma unroll
    for (int kk = 0; kk < 4; ++kk) {
      v16h bf = bfrag_lds(sWa, WA_STRIDE, col, kk * 32, lane);
      acc = __builtin_amdgcn_wmma_f32_16x16x32_f16(false, ax[kk], false, bf,
                                                   (short)0, acc, false, false);
    }
    // ReLU + park as f16 in this wave's private LDS slab (D layout: col=lc, row=r+8*half)
#pragma unroll
    for (int r = 0; r < 8; ++r) {
      float v = __builtin_fmaxf(acc[r], 0.0f);
      int m = r + ((lane & 16) ? 8 : 0);
      myH1[m * H1_STRIDE + col] = (_Float16)v;
    }
  }

  // ---- layer 2: h2 = relu(W_e2e @ h1 + b_e2e); weighted row-sum into accum[b] ----
  v16h ah[8];
#pragma unroll
  for (int kk = 0; kk < 8; ++kk) ah[kk] = afrag_lds(myH1, H1_STRIDE, lc, kk * 32, lane);

  float wrow[8];
#pragma unroll
  for (int r = 0; r < 8; ++r) {
    int n = ntile * MTILE + wave * 16 + r + ((lane & 16) ? 8 : 0);
    wrow[r] = (n < NODES) ? adj[n] : 0.0f;
  }

#pragma unroll 2
  for (int t = 0; t < 16; ++t) {
    const int col = t * 16 + lc;
    const float e2 = sB2[col];
    v8f acc;
#pragma unroll
    for (int r = 0; r < 8; ++r) acc[r] = e2;
#pragma unroll
    for (int kk = 0; kk < 8; ++kk) {
      v16h bf = bfrag_lds(sW2, W2_STRIDE, col, kk * 32, lane);
      acc = __builtin_amdgcn_wmma_f32_16x16x32_f16(false, ah[kk], false, bf,
                                                   (short)0, acc, false, false);
    }
    float partial = 0.0f;
#pragma unroll
    for (int r = 0; r < 8; ++r) {
      partial += wrow[r] * __builtin_fmaxf(acc[r], 0.0f);
    }
    partial += __shfl_xor(partial, 16, 32);       // fold the two lane-halves (same col)
    if (lane < 16) atomicAdd(&sAcc[col], partial);  // ds_add_f32
  }
  __syncthreads();
  unsafeAtomicAdd(&accum[b * HID + tid], sAcc[tid]);  // global_atomic_add_f32
}

// ---- tail: tiny per-batch MLP, scalar f32 ----
__global__ void __launch_bounds__(256) k_tail(const float* __restrict__ x,
                                              const float* __restrict__ accum,
                                              const float* __restrict__ We2n,
                                              const float* __restrict__ be2n,
                                              const float* __restrict__ Wn2n,
                                              const float* __restrict__ bn2n,
                                              const float* __restrict__ Wout,
                                              const float* __restrict__ bout,
                                              const int* __restrict__ ip,
                                              float* __restrict__ out) {
  __shared__ float s0[HID], s1[HID], s2[HID], xi[DIM];
  const int b = blockIdx.x, tid = threadIdx.x;
  const int i = *ip;
  s0[tid] = accum[b * HID + tid];
  if (tid < DIM) xi[tid] = x[((size_t)b * NODES + i) * DIM + tid];
  __syncthreads();
  float a = be2n[tid];
  for (int k = 0; k < HID; ++k) a += We2n[tid * HID + k] * s0[k];
  s1[tid] = __builtin_fmaxf(a, 0.0f);
  __syncthreads();
  float c = bn2n[tid];
  for (int k = 0; k < HID; ++k) c += Wn2n[tid * HID + k] * s1[k];
  s2[tid] = __builtin_fmaxf(c, 0.0f);
  __syncthreads();
  if (tid < DIM) {
    float o = bout[tid];
    const float* wr = Wout + tid * (DIM + HID);
    for (int k = 0; k < DIM; ++k) o += wr[k] * xi[k];
    for (int k = 0; k < HID; ++k) o += wr[DIM + k] * s2[k];
    out[b * DIM + tid] = o;
  }
}

extern "C" void kernel_launch(void* const* d_in, const int* in_sizes, int n_in,
                              void* d_out, int out_size, void* d_ws, size_t ws_size,
                              hipStream_t stream) {
  const float* x    = (const float*)d_in[0];
  const float* adj  = (const float*)d_in[1];
  const float* Wn2e = (const float*)d_in[2];
  const float* bn2e = (const float*)d_in[3];
  const float* We2e = (const float*)d_in[4];
  const float* be2e = (const float*)d_in[5];
  const float* We2n = (const float*)d_in[6];
  const float* be2n = (const float*)d_in[7];
  const float* Wn2n = (const float*)d_in[8];
  const float* bn2n = (const float*)d_in[9];
  const float* Wout = (const float*)d_in[10];
  const float* bout = (const float*)d_in[11];
  const int*   ip   = (const int*)d_in[12];
  float* out = (float*)d_out;

  char* ws = (char*)d_ws;
  _Float16* wah   = (_Float16*)(ws + 0);        // 256*128*2  = 65536 B
  _Float16* w2h   = (_Float16*)(ws + 65536);    // 256*256*2  = 131072 B
  float*    ebias = (float*)(ws + 196608);      // 512*256*4  = 524288 B
  float*    accum = (float*)(ws + 720896);      // 512*256*4  = 524288 B

  k_convert<<<256, 256, 0, stream>>>(Wn2e, We2e, wah, w2h);
  k_ebias<<<512, 256, 0, stream>>>(x, Wn2e, bn2e, ip, ebias, accum);

  // k_main needs ~269KB LDS (WGP has 320KB) -> dynamic shared mem
  (void)hipFuncSetAttribute(reinterpret_cast<const void*>(k_main),
                            hipFuncAttributeMaxDynamicSharedMemorySize, SMEM_BYTES);
  k_main<<<dim3(NTILES, 512), 256, SMEM_BYTES, stream>>>(x, adj, wah, w2h, ebias, be2e, accum);

  k_tail<<<512, 256, 0, stream>>>(x, accum, We2n, be2n, Wn2n, bn2n, Wout, bout, ip, out);
}
